// MoELayer_87832081203761
// MI455X (gfx1250) — matrix operand order repaired and checked by reading, not compiled, not run on hardware
//
#include <hip/hip_runtime.h>
#include <hip/hip_bf16.h>
#include <stdint.h>

// ---------------------------------------------------------------------------
// MoE (top-2 of 8 experts, SwiGLU FFN) for MI455X / gfx1250.
// fp32 -> bf16, v_wmma_f32_16x16x32_bf16 GEMMs, token-gather per expert.
// gateup: wave tile 32x32 (2x2, g+u fused -> 64 acc VGPRs, no spills)
// down:   wave tile 32x64 (2x4, single acc set -> 64 acc VGPRs)
// ---------------------------------------------------------------------------

typedef __bf16 bf16;
typedef bf16  v16bf __attribute__((ext_vector_type(16)));
typedef bf16  v8bf  __attribute__((ext_vector_type(8)));
typedef float v8f   __attribute__((ext_vector_type(8)));

#define T_TOK 8192
#define HDIM  1024
#define IDIM  2816
#define NEXP  8
#define CAP   8192          // max tokens per expert
#define AUXC  0.01f

// ---- workspace layout (bytes) ---------------------------------------------
static constexpr size_t OFF_CNT  = 0;                                  // int[8]
static constexpr size_t OFF_OFFS = 64;                                 // int[8]
static constexpr size_t OFF_PSUM = 128;                                // float[8]
static constexpr size_t OFF_TOK  = 256;                                // int[E*CAP]
static constexpr size_t OFF_WGT  = OFF_TOK + (size_t)NEXP*CAP*4;       // float[E*CAP]
static constexpr size_t OFF_XB   = 1u << 20;                           // bf16[T*H]
static constexpr size_t OFF_WGB  = OFF_XB  + (size_t)T_TOK*HDIM*2;     // bf16[E*I*H]
static constexpr size_t OFF_WUB  = OFF_WGB + (size_t)NEXP*IDIM*HDIM*2;
static constexpr size_t OFF_WDB  = OFF_WUB + (size_t)NEXP*IDIM*HDIM*2;
static constexpr size_t OFF_H    = OFF_WDB + (size_t)NEXP*HDIM*IDIM*2; // bf16[2T * I]
// total ~249 MB

// ---- WMMA helpers ----------------------------------------------------------
__device__ __forceinline__ v8f wmma_bf16(v16bf a, v16bf b, v8f c) {
  // D = A(16x32 bf16) x B(32x16 bf16) + C(16x16 f32)
  return __builtin_amdgcn_wmma_f32_16x16x32_bf16(false, a, false, b, (short)0, c, false, false);
}

// A fragment: lane holds row (lane&15); chunks at K = kb and kb+16, kb=(lane>>4)*8
__device__ __forceinline__ v16bf load_a(const bf16* __restrict__ row, int k0, int lane) {
  const int kb = (lane >> 4) * 8;
  v8bf lo = *(const v8bf*)(row + k0 + kb);
  v8bf hi = *(const v8bf*)(row + k0 + kb + 16);
  return __builtin_shufflevector(lo, hi, 0,1,2,3,4,5,6,7,8,9,10,11,12,13,14,15);
}
// B fragment: lane holds col (lane&15); chunks at K = kb and kb+16, kb=(lane>>4)*8
__device__ __forceinline__ v16bf load_b(const bf16* __restrict__ col, int k0, int lane) {
  const int kb = (lane >> 4) * 8;
  v8bf lo = *(const v8bf*)(col + k0 + kb);
  v8bf hi = *(const v8bf*)(col + k0 + kb + 16);
  return __builtin_shufflevector(lo, hi, 0,1,2,3,4,5,6,7,8,9,10,11,12,13,14,15);
}

__device__ __forceinline__ float fast_silu_mul(float g, float u) {
  // silu(g)*u with v_rcp_f32 instead of IEEE division
  return g * __builtin_amdgcn_rcpf(1.f + __expf(-g)) * u;
}

// ---- fp32 -> bf16 convert --------------------------------------------------
__global__ void cvt_kernel(const float* __restrict__ src, bf16* __restrict__ dst, int n) {
  int i = blockIdx.x * blockDim.x + threadIdx.x;
  const int stride = gridDim.x * blockDim.x;
  for (; i < n; i += stride) dst[i] = (bf16)src[i];
}

// ---- router: one wave per token -------------------------------------------
__global__ void router_kernel(const float* __restrict__ x, const float* __restrict__ gate_w,
                              int* __restrict__ cnt, float* __restrict__ psum,
                              int* __restrict__ tok, float* __restrict__ wgt) {
  __shared__ float gw[NEXP * HDIM];   // 32 KB of 320 KB LDS
  for (int i = threadIdx.x; i < NEXP * HDIM; i += blockDim.x) gw[i] = gate_w[i];
  __syncthreads();

  const int lane = threadIdx.x & 31;
  const int wid  = threadIdx.x >> 5;
  const int t    = blockIdx.x * (blockDim.x >> 5) + wid;
  if (t >= T_TOK) return;

  const float* xr = x + (size_t)t * HDIM;
  float acc[NEXP];
  #pragma unroll
  for (int e = 0; e < NEXP; ++e) acc[e] = 0.f;
  for (int k = lane; k < HDIM; k += 32) {
    const float xv = xr[k];
    #pragma unroll
    for (int e = 0; e < NEXP; ++e) acc[e] += xv * gw[e * HDIM + k];
  }
  #pragma unroll
  for (int e = 0; e < NEXP; ++e)
    #pragma unroll
    for (int off = 16; off > 0; off >>= 1) acc[e] += __shfl_xor(acc[e], off, 32);

  if (lane == 0) {
    float m = acc[0];
    #pragma unroll
    for (int e = 1; e < NEXP; ++e) m = fmaxf(m, acc[e]);
    float p[NEXP], s = 0.f;
    #pragma unroll
    for (int e = 0; e < NEXP; ++e) { p[e] = __expf(acc[e] - m); s += p[e]; }
    const float inv = 1.f / s;
    #pragma unroll
    for (int e = 0; e < NEXP; ++e) { p[e] *= inv; atomicAdd(&psum[e], p[e]); }
    int i0 = 0;
    #pragma unroll
    for (int e = 1; e < NEXP; ++e) if (p[e] > p[i0]) i0 = e;      // ties -> lowest idx
    int i1 = (i0 == 0) ? 1 : 0;
    #pragma unroll
    for (int e = 0; e < NEXP; ++e) if (e != i0 && p[e] > p[i1]) i1 = e;
    const float wn = 1.f / (p[i0] + p[i1]);
    const int s0 = atomicAdd(&cnt[i0], 1);
    tok[i0 * CAP + s0] = t;  wgt[i0 * CAP + s0] = p[i0] * wn;
    const int s1 = atomicAdd(&cnt[i1], 1);
    tok[i1 * CAP + s1] = t;  wgt[i1 * CAP + s1] = p[i1] * wn;
  }
}

// ---- finalize: prefix offsets + aux loss -----------------------------------
__global__ void finalize_kernel(const int* __restrict__ cnt, int* __restrict__ offs,
                                const float* __restrict__ psum, float* __restrict__ aux_out) {
  int o = 0;
  float a = 0.f;
  for (int e = 0; e < NEXP; ++e) {
    offs[e] = o;
    o += cnt[e];
    a += ((float)cnt[e] / (float)T_TOK) * (psum[e] / (float)T_TOK);
  }
  *aux_out = (float)NEXP * a * AUXC;
}

// ---- GEMM 1: h = silu(x Wg^T) * (x Wu^T), gathered per expert --------------
// block 256 = 8 waves (4x2); wave tile 32x32 (2x2, g+u); block tile 128x64
__global__ void __launch_bounds__(256)
moe_gateup(const bf16* __restrict__ xb, const bf16* __restrict__ Wgb,
           const bf16* __restrict__ Wub, const int* __restrict__ cnt,
           const int* __restrict__ offs, const int* __restrict__ tok,
           bf16* __restrict__ hbuf) {
  const int e  = blockIdx.z;
  const int cn = cnt[e];
  const int m0 = blockIdx.y * 128;
  if (m0 >= cn) return;
  const int base = offs[e];
  const int lane = threadIdx.x & 31;
  const int wid  = threadIdx.x >> 5;
  const int wm   = wid >> 1;                         // 0..3
  const int wn   = wid & 1;                          // 0..1
  const int n0   = blockIdx.x * 64 + wn * 32;        // I-column base for this wave
  const int lrow = lane & 15;

  const bf16* aptr[2];
  #pragma unroll
  for (int i = 0; i < 2; ++i) {
    int rr = m0 + wm * 32 + i * 16 + lrow;
    if (rr >= cn) rr = cn - 1;                        // clamp (stores are guarded)
    aptr[i] = xb + (size_t)tok[e * CAP + rr] * HDIM;
  }
  const bf16* bg[2]; const bf16* bu[2];
  #pragma unroll
  for (int j = 0; j < 2; ++j) {
    const size_t n = (size_t)(n0 + j * 16 + lrow);
    bg[j] = Wgb + (size_t)e * IDIM * HDIM + n * HDIM;
    bu[j] = Wub + (size_t)e * IDIM * HDIM + n * HDIM;
  }

  v8f accg[2][2] = {};
  v8f accu[2][2] = {};
  for (int k0 = 0; k0 < HDIM; k0 += 32) {
    v16bf a[2], Bg[2], Bu[2];
    #pragma unroll
    for (int i = 0; i < 2; ++i) a[i] = load_a(aptr[i], k0, lane);
    #pragma unroll
    for (int j = 0; j < 2; ++j) { Bg[j] = load_b(bg[j], k0, lane); Bu[j] = load_b(bu[j], k0, lane); }
    #pragma unroll
    for (int i = 0; i < 2; ++i)
      #pragma unroll
      for (int j = 0; j < 2; ++j) {
        accg[i][j] = wmma_bf16(a[i], Bg[j], accg[i][j]);
        accu[i][j] = wmma_bf16(a[i], Bu[j], accu[i][j]);
      }
  }

  const int lg = lane >> 4;
  #pragma unroll
  for (int i = 0; i < 2; ++i)
    #pragma unroll
    for (int v = 0; v < 8; ++v) {
      const int row = m0 + wm * 32 + i * 16 + lg * 8 + v;   // C layout: M = v + 8*(lane>>4)
      if (row < cn) {
        bf16* hrow = hbuf + (size_t)(base + row) * IDIM;
        #pragma unroll
        for (int j = 0; j < 2; ++j)
          hrow[n0 + j * 16 + lrow] = (bf16)fast_silu_mul(accg[i][j][v], accu[i][j][v]);
      }
    }
}

// ---- GEMM 2: out[token] += weight * (h Wd^T) -------------------------------
// block 256 = 8 waves (4x2); wave tile 32x64 (2x4); block tile 128x128
__global__ void __launch_bounds__(256)
moe_down(const bf16* __restrict__ hbuf, const bf16* __restrict__ Wdb,
         const int* __restrict__ cnt, const int* __restrict__ offs,
         const int* __restrict__ tok, const float* __restrict__ wgt,
         float* __restrict__ out) {
  const int e  = blockIdx.z;
  const int cn = cnt[e];
  const int m0 = blockIdx.y * 128;
  if (m0 >= cn) return;
  const int base = offs[e];
  const int lane = threadIdx.x & 31;
  const int wid  = threadIdx.x >> 5;
  const int wm   = wid >> 1;
  const int wn   = wid & 1;
  const int n0   = blockIdx.x * 128 + wn * 64;        // H-column base
  const int lrow = lane & 15;

  const bf16* aptr[2];
  #pragma unroll
  for (int i = 0; i < 2; ++i) {
    int rr = m0 + wm * 32 + i * 16 + lrow;
    if (rr >= cn) rr = cn - 1;
    aptr[i] = hbuf + (size_t)(base + rr) * IDIM;
  }
  const bf16* bptr[4];
  #pragma unroll
  for (int j = 0; j < 4; ++j) {
    const size_t n = (size_t)(n0 + j * 16 + lrow);
    bptr[j] = Wdb + (size_t)e * HDIM * IDIM + n * IDIM;
  }

  v8f acc[2][4] = {};
  for (int k0 = 0; k0 < IDIM; k0 += 32) {
    v16bf a[2], Bb[4];
    #pragma unroll
    for (int i = 0; i < 2; ++i) a[i]  = load_a(aptr[i], k0, lane);
    #pragma unroll
    for (int j = 0; j < 4; ++j) Bb[j] = load_b(bptr[j], k0, lane);
    #pragma unroll
    for (int i = 0; i < 2; ++i)
      #pragma unroll
      for (int j = 0; j < 4; ++j)
        acc[i][j] = wmma_bf16(a[i], Bb[j], acc[i][j]);
  }

  const int lg = lane >> 4;
  #pragma unroll
  for (int i = 0; i < 2; ++i)
    #pragma unroll
    for (int v = 0; v < 8; ++v) {
      const int row = m0 + wm * 32 + i * 16 + lg * 8 + v;
      if (row < cn) {
        const int   t = tok[e * CAP + row];
        const float w = wgt[e * CAP + row];
        float* orow = out + (size_t)t * HDIM;
        #pragma unroll
        for (int j = 0; j < 4; ++j)
          atomicAdd(&orow[n0 + j * 16 + lrow], w * acc[i][j][v]);
      }
    }
}

// ---------------------------------------------------------------------------
extern "C" void kernel_launch(void* const* d_in, const int* in_sizes, int n_in,
                              void* d_out, int out_size, void* d_ws, size_t ws_size,
                              hipStream_t stream) {
  (void)in_sizes; (void)n_in; (void)ws_size;
  const float* x  = (const float*)d_in[0];   // [B,S,H]
  const float* gw = (const float*)d_in[1];   // [E,H]
  const float* Wg = (const float*)d_in[2];   // [E,I,H]
  const float* Wu = (const float*)d_in[3];   // [E,I,H]
  const float* Wd = (const float*)d_in[4];   // [E,H,I]
  float* out = (float*)d_out;                // [B*S*H] + aux scalar
  char*  ws  = (char*)d_ws;

  int*   cnt  = (int*)  (ws + OFF_CNT);
  int*   offs = (int*)  (ws + OFF_OFFS);
  float* psum = (float*)(ws + OFF_PSUM);
  int*   tok  = (int*)  (ws + OFF_TOK);
  float* wgt  = (float*)(ws + OFF_WGT);
  bf16*  xb   = (bf16*) (ws + OFF_XB);
  bf16*  Wgb  = (bf16*) (ws + OFF_WGB);
  bf16*  Wub  = (bf16*) (ws + OFF_WUB);
  bf16*  Wdb  = (bf16*) (ws + OFF_WDB);
  bf16*  hbuf = (bf16*) (ws + OFF_H);

  hipMemsetAsync(d_out, 0, (size_t)out_size * sizeof(float), stream);
  hipMemsetAsync(ws, 0, 256, stream);   // cnt / offs / psum

  cvt_kernel<<<2048, 256, 0, stream>>>(x,  xb,  T_TOK * HDIM);
  cvt_kernel<<<4096, 256, 0, stream>>>(Wg, Wgb, NEXP * IDIM * HDIM);
  cvt_kernel<<<4096, 256, 0, stream>>>(Wu, Wub, NEXP * IDIM * HDIM);
  cvt_kernel<<<4096, 256, 0, stream>>>(Wd, Wdb, NEXP * HDIM * IDIM);

  router_kernel<<<T_TOK / 8, 256, 0, stream>>>(x, gw, cnt, psum, tok, wgt);
  finalize_kernel<<<1, 1, 0, stream>>>(cnt, offs, psum, out + (size_t)T_TOK * HDIM);

  dim3 g1(IDIM / 64, CAP / 128, NEXP);    // 44 x 64 x 8
  moe_gateup<<<g1, 256, 0, stream>>>(xb, Wgb, Wub, cnt, offs, tok, hbuf);

  dim3 g2(HDIM / 128, CAP / 128, NEXP);   // 8 x 64 x 8
  moe_down<<<g2, 256, 0, stream>>>(hbuf, Wdb, cnt, offs, tok, wgt, out);
}
